// Quantizer_10840497455530
// MI455X (gfx1250) — matrix operand order repaired and checked by reading, not compiled, not run on hardware
//
#include <hip/hip_runtime.h>
#include <hip/hip_bf16.h>

// ---------------------------------------------------------------------------
// VQ nearest-codebook lookup:
//   dists = -2 x@E^T + ||E||^2 ; idx = argmin_v ; out = E[idx]
// GEMM on v_wmma_f32_16x16x32_bf16 (hi/lo bf16 split, 3 products ~ fp32).
// Codebook tiles streamed into double-buffered LDS by the Tensor Data Mover
// (tensor_load_to_lds + s_wait_tensorcnt), overlapped with WMMA compute.
// ---------------------------------------------------------------------------

#define VQ_D   256              // feature dim
#define VQ_V   4096             // codebook size
#define VQ_KC  32               // K per bf16 WMMA
#define VQ_NK  (VQ_D / VQ_KC)   // 8 K-chunks
#define VQ_RW  (2 * VQ_D)       // packed row: 256 hi + 256 lo bf16 = 512 u16

typedef __attribute__((ext_vector_type(16))) __bf16          v16bf;
typedef __attribute__((ext_vector_type(16))) unsigned short  v16us;
typedef __attribute__((ext_vector_type(8)))  float           v8f;
typedef __attribute__((ext_vector_type(4)))  unsigned int    uint32x4;
typedef __attribute__((ext_vector_type(8)))  int             int32x8;
typedef __attribute__((ext_vector_type(4)))  int             int32x4;

#if defined(__has_builtin)
#if __has_builtin(__builtin_amdgcn_tensor_load_to_lds) && \
    __has_builtin(__builtin_amdgcn_s_wait_tensorcnt)
#define VQ_TDM 1
#endif
#endif
#ifndef VQ_TDM
#define VQ_TDM 0
#endif

__device__ __forceinline__ unsigned short f2bf(float f) {
    unsigned u = __builtin_bit_cast(unsigned, f);
    unsigned r = u + 0x7FFFu + ((u >> 16) & 1u);   // round-to-nearest-even
    return (unsigned short)(r >> 16);
}
__device__ __forceinline__ float bf2f(unsigned short h) {
    return __builtin_bit_cast(float, (unsigned)h << 16);
}

#if VQ_TDM
// Byte address of a __shared__ object in the LDS (addrspace-3) address space.
__device__ __forceinline__ unsigned lds_addr_of(const void* p) {
    return (unsigned)(unsigned long long)
        (__attribute__((address_space(3))) const void*)p;
}

// One TDM descriptor: 2D tile of 16 packed codebook rows (1 KB each) -> LDS.
// data_size = 8 bytes; tensor row = 128 units; tile = 128 x 16.
__device__ __forceinline__ void tdm_load_tile(const unsigned short* gsrc,
                                              unsigned lds_byte_addr) {
    unsigned long long ga = (unsigned long long)gsrc;
    uint32x4 g0 = { 1u,                                   // count=1 (valid D#)
                    lds_byte_addr,                        // lds_addr
                    (unsigned)ga,                         // global_addr[31:0]
                    (unsigned)(ga >> 32) | (2u << 30) };  // addr[56:32] | type=2
    int32x8 g1 = { (int)(3u << 16),        // data_size=3 (8B), no flags
                   (int)(128u << 16),      // tensor_dim0[15:0]=128 (bits 63:48)
                   (int)(4096u << 16),     // tensor_dim1[15:0]=4096 (bits 95:80)
                   (int)(128u << 16),      // tile_dim0=128 (bits 127:112)
                   16,                     // tile_dim1=16
                   128,                    // tensor_dim0_stride[31:0]=128
                   0, 0 };                 // stride hi / dim1_stride = 0
    int32x4 gz4 = { 0, 0, 0, 0 };          // groups 2/3 unused (2D tensor)
    int32x8 gz8 = { 0, 0, 0, 0, 0, 0, 0, 0 };
    // clang-23 / therock-10.0 form: (g0, g1, g2, g3, g_extra, cpol)
    __builtin_amdgcn_tensor_load_to_lds(g0, g1, gz4, gz4, gz8, 0);
}
#endif

// ---------------------------------------------------------------------------
// Prep: pack E rows as [256 bf16 hi | 256 bf16 lo] and compute ||E_v||^2.
// ---------------------------------------------------------------------------
__global__ __launch_bounds__(VQ_D) void vq_prep(const float* __restrict__ E,
                                                unsigned short* __restrict__ epk,
                                                float* __restrict__ norms) {
    __shared__ float red[VQ_D];
    const int v = blockIdx.x;
    const int t = threadIdx.x;

    float f = E[(size_t)v * VQ_D + t];
    unsigned short h = f2bf(f);
    epk[(size_t)v * VQ_RW + t]         = h;
    epk[(size_t)v * VQ_RW + VQ_D + t]  = f2bf(f - bf2f(h));

    red[t] = f * f;
    __syncthreads();
    for (int s = VQ_D / 2; s > 0; s >>= 1) {
        if (t < s) red[t] += red[t + s];
        __syncthreads();
    }
    if (t == 0) norms[v] = red[0];
}

// ---------------------------------------------------------------------------
// Main: 8 waves/block, each wave owns 16 x-rows; block shares double-buffered
// 16-code packed tiles in LDS, streamed by the TDM one tile ahead.
// ---------------------------------------------------------------------------
__global__ __launch_bounds__(256) void vq_main(const float* __restrict__ x,
                                               const float* __restrict__ E,
                                               const unsigned short* __restrict__ epk,
                                               const float* __restrict__ norms,
                                               float* __restrict__ out) {
    __shared__ __align__(32) unsigned short buf[2][16 * VQ_RW];   // 2 x 16 KB
    __shared__ int lds_idx[128];

    const int tid  = threadIdx.x;
    const int wave = tid >> 5;           // 0..7
    const int lane = tid & 31;
    const int l16  = lane & 15;
    const int hl   = lane >> 4;          // 0 or 1
    const size_t rowBase = (size_t)blockIdx.x * 128 + (size_t)wave * 16;

    // ---- A fragments: this wave's 16 x-rows, bf16 hi/lo, contiguous K order, same as B
    //      (lanes 0-15: K 0-15 ; lanes 16-31: K 16-31).
    v16us a_hi[VQ_NK], a_lo[VQ_NK];
    {
        const float* xr = x + (rowBase + (size_t)l16) * VQ_D;
        for (int c = 0; c < VQ_NK; ++c) {
            const int k0 = c * VQ_KC;
#pragma unroll
            for (int j = 0; j < 8; ++j) {
                float f0 = xr[k0 + hl * 16 + j];
                float f1 = xr[k0 + hl * 16 + 8 + j];
                unsigned short h0 = f2bf(f0);
                unsigned short h1 = f2bf(f1);
                a_hi[c][j]     = h0;
                a_hi[c][8 + j] = h1;
                a_lo[c][j]     = f2bf(f0 - bf2f(h0));
                a_lo[c][8 + j] = f2bf(f1 - bf2f(h1));
            }
        }
    }

    float best[8];
    int   bidx[8];
#pragma unroll
    for (int j = 0; j < 8; ++j) { best[j] = 3.4e38f; bidx[j] = 0; }

    // ---- Prologue: bring tile 0 into buf[0].
#if VQ_TDM
    if (wave == 0) {
        tdm_load_tile(epk, lds_addr_of(&buf[0][0]));
        __builtin_amdgcn_s_wait_tensorcnt(0);
    }
#else
    {
        const uint4* g = (const uint4*)epk;
        uint4* s = (uint4*)&buf[0][0];
        for (int i = tid; i < 1024; i += 256) s[i] = g[i];
    }
#endif
    __syncthreads();

    for (int vb = 0; vb < VQ_V; vb += 16) {
        const int p = (vb >> 4) & 1;
        const unsigned short* cur = &buf[p][0];

        // ---- Kick off next tile into the other buffer (overlaps compute).
#if VQ_TDM
        if (wave == 0 && vb + 16 < VQ_V)
            tdm_load_tile(epk + (size_t)(vb + 16) * VQ_RW,
                          lds_addr_of(&buf[p ^ 1][0]));
#else
        uint4 stage[4];
        if (vb + 16 < VQ_V) {
            const uint4* g = (const uint4*)(epk + (size_t)(vb + 16) * VQ_RW);
#pragma unroll
            for (int q = 0; q < 4; ++q) stage[q] = g[tid + q * 256];
        }
#endif

        // ---- 16x16 dot tile: acc0 = hi*hi, acc1 = hi*lo + lo*hi.
        v8f acc0 = {};
        v8f acc1 = {};
#pragma unroll
        for (int c = 0; c < VQ_NK; ++c) {
            // B fragment: column n = l16 (code vb+l16), K start = hl*16.
            const unsigned short* bh = cur + l16 * VQ_RW + c * VQ_KC + hl * 16;
            const unsigned short* bl = bh + VQ_D;
            v16bf bhi = __builtin_bit_cast(v16bf, *(const v16us*)bh);
            v16bf blo = __builtin_bit_cast(v16bf, *(const v16us*)bl);
            v16bf ahi = __builtin_bit_cast(v16bf, a_hi[c]);
            v16bf alo = __builtin_bit_cast(v16bf, a_lo[c]);
            acc0 = __builtin_amdgcn_wmma_f32_16x16x32_bf16(false, ahi, false, bhi,
                                                           (short)0, acc0, false, false);
            acc1 = __builtin_amdgcn_wmma_f32_16x16x32_bf16(false, ahi, false, blo,
                                                           (short)0, acc1, false, false);
            acc1 = __builtin_amdgcn_wmma_f32_16x16x32_bf16(false, alo, false, bhi,
                                                           (short)0, acc1, false, false);
        }

        // ---- dist = ||E_v||^2 - 2*dot ; running per-lane argmin.
        const float nrm = norms[vb + l16];
#pragma unroll
        for (int j = 0; j < 8; ++j) {
            float dist = nrm - 2.0f * (acc0[j] + acc1[j]);
            if (dist < best[j]) { best[j] = dist; bidx[j] = vb + l16; }
        }

        // ---- Pipeline handoff: next tile resident, all waves done with cur.
#if VQ_TDM
        if (wave == 0) __builtin_amdgcn_s_wait_tensorcnt(0);
#else
        if (vb + 16 < VQ_V) {
            uint4* s = (uint4*)&buf[p ^ 1][0];
#pragma unroll
            for (int q = 0; q < 4; ++q) s[tid + q * 256] = stage[q];
        }
#endif
        __syncthreads();
    }

    // ---- Reduce (min,idx) across the 16 lane-columns of each half-wave.
#pragma unroll
    for (int j = 0; j < 8; ++j) {
        float b = best[j];
        int   i = bidx[j];
        for (int m = 8; m >= 1; m >>= 1) {
            float ob = __shfl_xor(b, m, 16);
            int   oi = __shfl_xor(i, m, 16);
            if (ob < b) { b = ob; i = oi; }
        }
        best[j] = b; bidx[j] = i;
    }
    // Lane 0 holds rows 0-7, lane 16 holds rows 8-15 of the wave's tile.
    if (l16 == 0) {
#pragma unroll
        for (int j = 0; j < 8; ++j) lds_idx[wave * 16 + hl * 8 + j] = bidx[j];
    }
    __syncthreads();

    // ---- Gather epilogue: out[row] = E[idx[row]] in fp32 (float4 copies).
    for (int i = tid; i < 128 * (VQ_D / 4); i += 256) {
        const int r  = i / (VQ_D / 4);
        const int kk = (i % (VQ_D / 4)) * 4;
        const int idx = lds_idx[r];
        const float4 srcv = *(const float4*)(E + (size_t)idx * VQ_D + kk);
        *(float4*)(out + ((size_t)blockIdx.x * 128 + r) * VQ_D + kk) = srcv;
    }
}

// ---------------------------------------------------------------------------
extern "C" void kernel_launch(void* const* d_in, const int* in_sizes, int n_in,
                              void* d_out, int out_size, void* d_ws, size_t ws_size,
                              hipStream_t stream) {
    const float* x = (const float*)d_in[0];   // [32,1024,256] fp32
    const float* E = (const float*)d_in[1];   // [4096,256]    fp32
    float* out = (float*)d_out;

    const int N = in_sizes[0] / VQ_D;         // 32768 rows

    // Workspace: packed hi|lo codebook (4 MB) | norms (16 KB)
    unsigned short* epk = (unsigned short*)d_ws;
    float* norms = (float*)(epk + (size_t)VQ_V * VQ_RW);

    vq_prep<<<VQ_V, VQ_D, 0, stream>>>(E, epk, norms);
    vq_main<<<N / 128, 256, 0, stream>>>(x, E, epk, norms, out);
}
